// MHA_38603166057190
// MI455X (gfx1250) — matrix operand order repaired and checked by reading, not compiled, not run on hardware
//
#include <hip/hip_runtime.h>

// ---------------------------------------------------------------------------
// Linformer MHA for MI455X (gfx1250, wave32, WMMA 16x16x32 f16 -> f32 acc)
//
// Pipeline (all f16 intermediates, conversions fused into LDS staging):
//   1) k_ds[b] = kds(64x512)  @ key[b](512x768)        (batched GEMM, shared A)
//      v_ds[b] = vds(64x512)  @ value[b](512x768)
//   2) q  = query(32768x768)  @ Wq(768x768) + bq
//      k  = k_ds (4096x768)   @ Wk(768x768) + bk
//      v  = v_ds (4096x768)   @ Wv(768x768) + bv
//   3) per (b, head, 128-row S chunk): scores WMMA, masked softmax, attn@V WMMA
//   4) out = a_v(32768x768)   @ Wo(768x768) + bo       (f32 output)
//
// GEMM mainloop is double-buffered; f16 A tiles are staged with
// GLOBAL_LOAD_ASYNC_TO_LDS_B128 (ASYNCcnt) so the copy overlaps the WMMAs.
// ---------------------------------------------------------------------------

typedef __attribute__((ext_vector_type(16))) _Float16 v16h;
typedef __attribute__((ext_vector_type(8)))  float    v8f;
typedef __attribute__((ext_vector_type(4)))  int      v4i;

union Frag { uint4 u[2]; v16h v; };
union H4   { _Float16 h[4]; uint2 u; };
union H8   { uint4 u; _Float16 h[8]; };

__device__ __forceinline__ v8f wmma16(const Frag& a, const Frag& b, v8f c) {
  // (neg_a, A, neg_b, B, c_mod, C, reuse_a, reuse_b)
  return __builtin_amdgcn_wmma_f32_16x16x32_f16(false, a.v, false, b.v,
                                                (short)0, c, false, false);
}

// ---- CDNA5 async global->LDS copy (ASYNCcnt-tracked) ----------------------
// Builtin signature (from hipcc diagnostic): param0 = v4i in AS(1), non-const.
__device__ __forceinline__ void async_b128(void* lds, const void* g) {
#if __has_builtin(__builtin_amdgcn_global_load_async_to_lds_b128)
  void* gg = const_cast<void*>(g);
  __builtin_amdgcn_global_load_async_to_lds_b128(
      (__attribute__((address_space(1))) v4i*)gg,
      (__attribute__((address_space(3))) v4i*)lds, 0, 0);
#else
  unsigned l = (unsigned)(unsigned long long)lds;   // low 32 bits = LDS offset
  asm volatile("global_load_async_to_lds_b128 %0, %1, off"
               :: "v"(l), "v"(g) : "memory");
#endif
}

__device__ __forceinline__ void wait_async0() {
#if __has_builtin(__builtin_amdgcn_s_wait_asynccnt)
  __builtin_amdgcn_s_wait_asynccnt(0);
#else
  asm volatile("s_wait_asynccnt 0x0" ::: "memory");
#endif
}

// ---------------------------------------------------------------------------
// Tiled GEMM: C[M,N] = A[M,Kd] * B[Kd,N] (+bias), A f32-or-f16, B f32,
// C f32-or-f16. BM=BN=128, BK=32, double-buffered. 256 thr = 8 waves arranged
// 4(M) x 2(N); each wave owns a 32x64 slab = 2x4 WMMA tiles. B is staged
// transposed (Bt[n][k]) so fragment loads are contiguous ds_load_b128.
// ---------------------------------------------------------------------------
template<bool A_F32, bool OUT_F32>
__global__ __launch_bounds__(256)
void gemm_f16_wmma(const void* __restrict__ A_, const float* __restrict__ B,
                   void* __restrict__ C_, const float* __restrict__ bias,
                   int M, int N, int Kd,
                   long long sA, long long sB, long long sC)
{
  __shared__ _Float16 As[2][128][40];   // [buf][row][k]  (pad 8 halfs)
  __shared__ _Float16 Bt[2][128][40];   // [buf][col][k]  (pad 8 halfs)

  const int tid  = threadIdx.x;
  const int lane = tid & 31;
  const int wv   = tid >> 5;
  const int wm   = wv >> 1;          // 0..3 : 32-row slab
  const int wn   = wv & 1;           // 0..1 : 64-col slab
  const int g    = lane >> 4;        // half-wave group
  const int lr   = lane & 15;

  const int mBase = blockIdx.y * 128;
  const int nBase = blockIdx.x * 128;
  const long long bz = blockIdx.z;

  const float*    Af = (const float*)A_    + bz * sA;
  const _Float16* Ah = (const _Float16*)A_ + bz * sA;
  const float*    Bb = B + bz * sB;

  const v8f zero = {0.f,0.f,0.f,0.f,0.f,0.f,0.f,0.f};
  v8f acc[2][4];
  #pragma unroll
  for (int t = 0; t < 2; ++t)
    #pragma unroll
    for (int u = 0; u < 4; ++u) acc[t][u] = zero;

  float4 bReg[4];          // B tile in flight (f32, converted on commit)
  float4 aReg[4];          // A tile in flight (f32 path only)

  // fetch tile at k0: f16 A goes async straight into LDS buf; f32 A and B
  // land in registers (conversion fused on commit).
  auto prefetch = [&](int k0, int buf) {
    if constexpr (A_F32) {
      #pragma unroll
      for (int i = 0; i < 4; ++i) {
        int idx = tid + i * 256;                 // 1024 float4 chunks
        int r = idx >> 3, c4 = (idx & 7) * 4;
        int gr = mBase + r; if (gr > M - 1) gr = M - 1;
        aReg[i] = *(const float4*)(Af + (long long)gr * Kd + k0 + c4);
      }
    } else {
      #pragma unroll
      for (int i = 0; i < 2; ++i) {
        int idx = tid + i * 256;                 // 512 b128 chunks
        int r = idx >> 2, c8 = (idx & 3) * 8;
        int gr = mBase + r; if (gr > M - 1) gr = M - 1;
        async_b128(&As[buf][r][c8], Ah + (long long)gr * Kd + k0 + c8);
      }
    }
    #pragma unroll
    for (int i = 0; i < 4; ++i) {
      int idx = tid + i * 256;                   // 1024 float4 chunks
      int kk = idx >> 5, c4 = (idx & 31) * 4;
      bReg[i] = *(const float4*)(Bb + (long long)(k0 + kk) * N + nBase + c4);
    }
  };

  // convert + store the register-staged parts of the tile into LDS buf
  auto commit = [&](int buf) {
    if constexpr (A_F32) {
      #pragma unroll
      for (int i = 0; i < 4; ++i) {
        int idx = tid + i * 256;
        int r = idx >> 3, c4 = (idx & 7) * 4;
        H4 p; p.h[0] = (_Float16)aReg[i].x; p.h[1] = (_Float16)aReg[i].y;
              p.h[2] = (_Float16)aReg[i].z; p.h[3] = (_Float16)aReg[i].w;
        *(uint2*)&As[buf][r][c4] = p.u;
      }
    }
    #pragma unroll
    for (int i = 0; i < 4; ++i) {
      int idx = tid + i * 256;
      int kk = idx >> 5, c4 = (idx & 31) * 4;    // Bt[n][k] = B[k][n]
      Bt[buf][c4 + 0][kk] = (_Float16)bReg[i].x;
      Bt[buf][c4 + 1][kk] = (_Float16)bReg[i].y;
      Bt[buf][c4 + 2][kk] = (_Float16)bReg[i].z;
      Bt[buf][c4 + 3][kk] = (_Float16)bReg[i].w;
    }
  };

  const int nIter = Kd >> 5;
  prefetch(0, 0);
  for (int it = 0; it < nIter; ++it) {
    const int cur = it & 1;
    commit(cur);
    if constexpr (!A_F32) wait_async0();   // this wave's async A(cur) landed
    __syncthreads();                       // tile cur visible block-wide
    if (it + 1 < nIter) prefetch((it + 1) * 32, cur ^ 1);  // overlaps WMMAs

    Frag a[2], bf[4];
    #pragma unroll
    for (int t = 0; t < 2; ++t) {
      int r = wm * 32 + t * 16 + lr;             // A row for this lane
      a[t].u[0] = *(const uint4*)&As[cur][r][g * 8];        // K = 8g + 0..7
      a[t].u[1] = *(const uint4*)&As[cur][r][g * 8 + 16];   // K = 16+8g+0..7
    }
    #pragma unroll
    for (int u = 0; u < 4; ++u) {
      int n = wn * 64 + u * 16 + lr;             // B column for this lane
      bf[u].u[0] = *(const uint4*)&Bt[cur][n][g * 8];       // K = 8g + 0..7
      bf[u].u[1] = *(const uint4*)&Bt[cur][n][g * 8 + 16];  // K = 16+8g+0..7
    }
    #pragma unroll
    for (int t = 0; t < 2; ++t)
      #pragma unroll
      for (int u = 0; u < 4; ++u)
        acc[t][u] = wmma16(a[t], bf[u], acc[t][u]);
    __syncthreads();                       // all readers done with buf cur
  }

  // ---- writeback (+bias) ----
  float bias4[4];
  #pragma unroll
  for (int u = 0; u < 4; ++u) {
    int col = nBase + wn * 64 + u * 16 + lr;
    bias4[u] = bias ? bias[col] : 0.f;
  }
  #pragma unroll
  for (int t = 0; t < 2; ++t)
    #pragma unroll
    for (int u = 0; u < 4; ++u)
      #pragma unroll
      for (int j = 0; j < 8; ++j) {
        int row = mBase + wm * 32 + t * 16 + g * 8 + j;
        int col = nBase + wn * 64 + u * 16 + lr;
        if (row < M) {
          float val = acc[t][u][j] + bias4[u];
          if constexpr (OUT_F32)
            ((float*)C_)[bz * sC + (long long)row * N + col] = val;
          else
            ((_Float16*)C_)[bz * sC + (long long)row * N + col] = (_Float16)val;
        }
      }
}

// ---------------------------------------------------------------------------
// Fused attention per (s-chunk of 128, head n, batch b).
// 256 thr = 8 waves; wave w owns rows w*16..w*16+15 of the chunk.
// scores(16x64) via WMMA straight from global f16 q/k (layouts already match
// fragment shapes); masked stable softmax via LDS + half-group shfl; then
// attn(16x64) @ v_head(64x64) via WMMA against LDS-transposed V.
// ---------------------------------------------------------------------------
__global__ __launch_bounds__(256)
void attn_wmma(const _Float16* __restrict__ q, const _Float16* __restrict__ k,
               const _Float16* __restrict__ v, _Float16* __restrict__ av)
{
  __shared__ _Float16 vT[64][72];       // vT[h][kk], pad 8
  __shared__ float    sc[8][16][64];    // per-wave score scratch
  __shared__ _Float16 at[8][16][64];    // per-wave attn weights (f16)

  const int tid  = threadIdx.x;
  const int lane = tid & 31;
  const int w    = tid >> 5;
  const int g    = lane >> 4;
  const int lr   = lane & 15;
  const int sBase = blockIdx.x * 128;
  const int n     = blockIdx.y;
  const long long b = blockIdx.z;

  // stage + transpose V head: v[b,kk,n,h] -> vT[h][kk]
  #pragma unroll
  for (int i = 0; i < 2; ++i) {
    int idx = tid + i * 256;
    int kk = idx >> 3, h0 = (idx & 7) * 8;
    H8 t; t.u = *(const uint4*)(v + (((b * 64 + kk) * 12 + n) << 6) + h0);
    #pragma unroll
    for (int j = 0; j < 8; ++j) vT[h0 + j][kk] = t.h[j];
  }
  __syncthreads();

  const v8f zero = {0.f,0.f,0.f,0.f,0.f,0.f,0.f,0.f};

  // -------- scores = q_rows(16xH=64) @ k_head^T (H contracted, 2 k-steps) ----
  v8f acc[4] = {zero, zero, zero, zero};
  const int rowg = sBase + w * 16 + lr;                      // this lane's q row
  const _Float16* qrow = q + (((b * 512 + rowg) * 12 + n) << 6);
  #pragma unroll
  for (int ks = 0; ks < 2; ++ks) {
    Frag aq;
    aq.u[0] = *(const uint4*)(qrow + ks * 32 + g * 8);
    aq.u[1] = *(const uint4*)(qrow + ks * 32 + g * 8 + 16);
    #pragma unroll
    for (int u = 0; u < 4; ++u) {
      int kkcol = u * 16 + lr;                               // score column
      const _Float16* kr = k + (((b * 64 + kkcol) * 12 + n) << 6) + ks * 32 + g * 8;
      Frag bk;
      bk.u[0] = *(const uint4*)(kr);
      bk.u[1] = *(const uint4*)(kr + 16);
      acc[u] = wmma16(aq, bk, acc[u]);
    }
  }

  // spill scores to this wave's LDS slab
  #pragma unroll
  for (int u = 0; u < 4; ++u)
    #pragma unroll
    for (int j = 0; j < 8; ++j)
      sc[w][g * 8 + j][u * 16 + lr] = acc[u][j];

  // -------- masked, numerically-stable softmax over 64 cols ----------------
  // lane handles row (lane&15), column half g; pair (l, l^16) combines.
  {
    const int row = lr;
    const int sg  = sBase + w * 16 + row;                    // global query pos
    float m = -3.4e38f;
    #pragma unroll
    for (int i = 0; i < 32; ++i) {
      int c = g * 32 + i;
      float x = sc[w][row][c];
      if (c > sg) x = -9e15f;                                // rectangular tril
      sc[w][row][c] = x;
      m = fmaxf(m, x);
    }
    m = fmaxf(m, __shfl_xor(m, 16, 32));
    float s = 0.f;
    #pragma unroll
    for (int i = 0; i < 32; ++i) {
      int c = g * 32 + i;
      float e = __expf(sc[w][row][c] - m);
      sc[w][row][c] = e;
      s += e;
    }
    s += __shfl_xor(s, 16, 32);
    float inv = 1.f / s;
    #pragma unroll
    for (int i = 0; i < 32; ++i) {
      int c = g * 32 + i;
      at[w][row][c] = (_Float16)(sc[w][row][c] * inv);
    }
  }

  // -------- a_v = attn(16x64) @ v_head(64x64), K contracted ----------------
  v8f acc2[4] = {zero, zero, zero, zero};
  #pragma unroll
  for (int ks = 0; ks < 2; ++ks) {
    Frag aa;
    aa.u[0] = *(const uint4*)&at[w][lr][ks * 32 + g * 8];
    aa.u[1] = *(const uint4*)&at[w][lr][ks * 32 + g * 8 + 16];
    #pragma unroll
    for (int u = 0; u < 4; ++u) {
      Frag bv_;
      bv_.u[0] = *(const uint4*)&vT[u * 16 + lr][ks * 32 + g * 8];
      bv_.u[1] = *(const uint4*)&vT[u * 16 + lr][ks * 32 + g * 8 + 16];
      acc2[u] = wmma16(aa, bv_, acc2[u]);
    }
  }

  // write a_v[b,s,n,h] (f16, consumed by the output-projection GEMM)
  #pragma unroll
  for (int u = 0; u < 4; ++u)
    #pragma unroll
    for (int j = 0; j < 8; ++j) {
      int s = sBase + w * 16 + g * 8 + j;
      int h = u * 16 + lr;
      av[(((b * 512 + s) * 12 + n) << 6) + h] = (_Float16)acc2[u][j];
    }
}

// ---------------------------------------------------------------------------
extern "C" void kernel_launch(void* const* d_in, const int* in_sizes, int n_in,
                              void* d_out, int out_size, void* d_ws, size_t ws_size,
                              hipStream_t stream)
{
  const float* query = (const float*)d_in[0];
  const float* key   = (const float*)d_in[1];
  const float* value = (const float*)d_in[2];
  const float* kds   = (const float*)d_in[3];
  const float* vds   = (const float*)d_in[4];
  const float* Wq    = (const float*)d_in[5];
  const float* bq    = (const float*)d_in[6];
  const float* Wk    = (const float*)d_in[7];
  const float* bk    = (const float*)d_in[8];
  const float* Wv    = (const float*)d_in[9];
  const float* bv    = (const float*)d_in[10];
  const float* Wo    = (const float*)d_in[11];
  const float* bo    = (const float*)d_in[12];

  // workspace: all f16 intermediates (~126 MB total)
  const long long KD = 64LL * 64 * 768;    // per-tensor downsampled size
  const long long QH = 64LL * 512 * 768;   // full-sequence size
  _Float16* ws  = (_Float16*)d_ws;
  _Float16* kd_h = ws;  ws += KD;          // k_ds  [B,64,768]
  _Float16* vd_h = ws;  ws += KD;          // v_ds  [B,64,768]
  _Float16* q_h  = ws;  ws += QH;          // q     [B,S,N,H]
  _Float16* k_h  = ws;  ws += KD;          // k     [B,64,N,H]
  _Float16* v_h  = ws;  ws += KD;          // v     [B,64,N,H]
  _Float16* av_h = ws;  ws += QH;          // a_v   [B,S,N,H]

  dim3 blk(256);

  // 1) Linformer downsample (batched; A=kds shared across batch)
  gemm_f16_wmma<true,  false><<<dim3(6, 1, 64), blk, 0, stream>>>(
      kds, key,   kd_h, nullptr, 64, 768, 512, 0LL, 512LL * 768, 64LL * 768);
  gemm_f16_wmma<true,  false><<<dim3(6, 1, 64), blk, 0, stream>>>(
      vds, value, vd_h, nullptr, 64, 768, 512, 0LL, 512LL * 768, 64LL * 768);

  // 2) QKV projections
  gemm_f16_wmma<true,  false><<<dim3(6, 256, 1), blk, 0, stream>>>(
      query, Wq, q_h, bq, 32768, 768, 768, 0LL, 0LL, 0LL);
  gemm_f16_wmma<false, false><<<dim3(6, 32, 1), blk, 0, stream>>>(
      kd_h,  Wk, k_h, bk,  4096, 768, 768, 0LL, 0LL, 0LL);
  gemm_f16_wmma<false, false><<<dim3(6, 32, 1), blk, 0, stream>>>(
      vd_h,  Wv, v_h, bv,  4096, 768, 768, 0LL, 0LL, 0LL);

  // 3) fused masked-softmax attention per (s-chunk, head, batch)
  attn_wmma<<<dim3(4, 12, 64), blk, 0, stream>>>(q_h, k_h, v_h, av_h);

  // 4) output projection (f32 out + bo)
  gemm_f16_wmma<false, true><<<dim3(6, 256, 1), blk, 0, stream>>>(
      av_h, Wo, d_out, bo, 32768, 768, 768, 0LL, 0LL, 0LL);
}
